// Seq2Seq_4990751998109
// MI455X (gfx1250) — matrix operand order repaired and checked by reading, not compiled, not run on hardware
//
#include <hip/hip_runtime.h>

typedef __attribute__((ext_vector_type(16))) _Float16 v16h;
typedef __attribute__((ext_vector_type(8)))  float    v8f;

#define H     300
#define DDIM  300
#define G4    1200      // 4*H
#define TSEQ  1024
#define TOUT  256

#define KT_ENC 10       // ceil(300/32) K-tiles (k padded to 320)
#define KT_DEC 20       // K = [h(0..299) pad..319 | x(320..619) pad..639]
#define MT_G   75       // 1200/16 gate row tiles
#define MT_FC  19       // ceil(300/16) fc row tiles
#define MT_X   64       // 1024/16 timestep tiles
#define NT_W   75       // 1200/16 gate col tiles

#define CACHE_TILES 288 // 288 KB of weight tiles resident in LDS (320 KB WGP LDS)

// ---------------------------------------------------------------------------
// Fragment packing.  Per-tile storage: [tile][lane(32)][elem(16)] f16 = 1 KB.
// A-matrix 16x32 f16 lane layout (contiguous K, same as B):
//   lane<16 : row=lane,    K = e
//   lane>=16: row=lane-16, K = e+16
// B-matrix 32x16 f16 lane layout: col = lane&15, K = (lane<16?0:16)+e
// ---------------------------------------------------------------------------
__global__ void pack_a(const float* __restrict__ src, _Float16* __restrict__ dst,
                       int Mtiles, int Ktiles, int R, int C) {
  int total = Mtiles * Ktiles * 512;
  for (int g = blockIdx.x * blockDim.x + threadIdx.x; g < total;
       g += gridDim.x * blockDim.x) {
    int tile = g >> 9, t = g & 511;
    int lane = t >> 4, e = t & 15;
    int mt = tile / Ktiles, kt = tile - mt * Ktiles;
    int m  = mt * 16 + (lane & 15);
    int j  = e & 7;
    int kk = ((lane < 16) ? 0 : 16) + e;
    int k  = kt * 32 + kk;
    float v = (m < R && k < C) ? src[(size_t)m * C + k] : 0.f;
    dst[g] = (_Float16)v;
  }
}

// A-fragments for K-concatenated [s0 | s1]: k in [0,320) -> s0, [320,640) -> s1
__global__ void pack_a2(const float* __restrict__ s0, int C0,
                        const float* __restrict__ s1, int C1,
                        _Float16* __restrict__ dst, int Mtiles, int Ktiles, int R) {
  int total = Mtiles * Ktiles * 512;
  for (int g = blockIdx.x * blockDim.x + threadIdx.x; g < total;
       g += gridDim.x * blockDim.x) {
    int tile = g >> 9, t = g & 511;
    int lane = t >> 4, e = t & 15;
    int mt = tile / Ktiles, kt = tile - mt * Ktiles;
    int m  = mt * 16 + (lane & 15);
    int j  = e & 7;
    int kk = ((lane < 16) ? 0 : 16) + e;
    int k  = kt * 32 + kk;
    float v = 0.f;
    if (m < R) {
      if (k < 320) { if (k < C0) v = s0[(size_t)m * C0 + k]; }
      else         { int k2 = k - 320; if (k2 < C1) v = s1[(size_t)m * C1 + k2]; }
    }
    dst[g] = (_Float16)v;
  }
}

// B-fragments of src^T:  B[k][n] = src[n][k]   (src is R x C row-major)
__global__ void pack_b(const float* __restrict__ src, _Float16* __restrict__ dst,
                       int Ktiles, int Ntiles, int R, int C) {
  int total = Ktiles * Ntiles * 512;
  for (int g = blockIdx.x * blockDim.x + threadIdx.x; g < total;
       g += gridDim.x * blockDim.x) {
    int tile = g >> 9, t = g & 511;
    int lane = t >> 4, e = t & 15;
    int kt = tile / Ntiles, nt = tile - kt * Ntiles;
    int n  = nt * 16 + (lane & 15);
    int kk = ((lane < 16) ? 0 : 16) + e;
    int k  = kt * 32 + kk;
    float v = (n < R && k < C) ? src[(size_t)n * C + k] : 0.f;
    dst[g] = (_Float16)v;
  }
}

// ---------------------------------------------------------------------------
// Gx[t][j] = X[t,:] . Wih[j,:] + bih[j] + bhh[j]   (1024 x 1200, K=300)
// One 16x16 output tile per wave, 10 chained WMMAs over K.
// ---------------------------------------------------------------------------
__global__ __launch_bounds__(256)
void enc_input_gemm(const _Float16* __restrict__ Xa, const _Float16* __restrict__ Wb,
                    const float* __restrict__ bih, const float* __restrict__ bhh,
                    float* __restrict__ Gx) {
  int wave = threadIdx.x >> 5;
  int lane = threadIdx.x & 31;
  int tileId = blockIdx.x * 8 + wave;
  if (tileId >= MT_X * NT_W) return;          // wave-uniform
  int mt = tileId / NT_W, nt = tileId - mt * NT_W;
  v8f acc = {};
#pragma unroll
  for (int kt = 0; kt < KT_ENC; ++kt) {
    v16h a = *(const v16h*)(Xa + (((size_t)(mt * KT_ENC + kt)) << 9) + (lane << 4));
    v16h b = *(const v16h*)(Wb + (((size_t)(kt * NT_W + nt)) << 9) + (lane << 4));
    acc = __builtin_amdgcn_wmma_f32_16x16x32_f16(false, a, false, b,
                                                 (short)0, acc, false, false);
  }
  int col  = nt * 16 + (lane & 15);
  float bias = bih[col] + bhh[col];
  int r0   = mt * 16 + ((lane >> 4) << 3);
#pragma unroll
  for (int i = 0; i < 8; ++i)
    Gx[(size_t)(r0 + i) * G4 + col] = acc[i] + bias;
}

// ---------------------------------------------------------------------------
// Persistent recurrence: one workgroup (32 wave32 waves) runs encoder+decoder.
// A = packed weight tiles: first CACHE_TILES resident in LDS, rest streamed
// from L2.  B = h/x broadcast into all 16 columns, built from f16 LDS image.
// ---------------------------------------------------------------------------
__device__ __forceinline__ float sig(float x) { return 1.f / (1.f + __expf(-x)); }

extern __shared__ __attribute__((aligned(32))) _Float16 tileCache[]; // CACHE_TILES KB

__device__ __forceinline__ void fill_cache(const _Float16* __restrict__ src,
                                           int ntiles) {
  int total16 = ntiles * 64;                    // 16-byte chunks (64 per 1KB tile)
  const int4* s = (const int4*)src;
  int4*       d = (int4*)tileCache;
  for (int i = threadIdx.x; i < total16; i += blockDim.x) d[i] = s[i];
}

__device__ __forceinline__ v16h load_a(const _Float16* __restrict__ glob, int tile,
                                       int lane) {
  if (tile < CACHE_TILES)                       // wave-uniform branch
    return *(const v16h*)(tileCache + (((size_t)tile) << 9) + (lane << 4));
  return *(const v16h*)(glob + (((size_t)tile) << 9) + (lane << 4));
}

__global__ __launch_bounds__(1024)
void seq_recurrence(const _Float16* __restrict__ WhhA,   // [75][10] tiles (encoder)
                    const _Float16* __restrict__ decA,   // [75][20] tiles ([Whh|Wih])
                    const _Float16* __restrict__ fcA,    // [19][10] tiles
                    const float* __restrict__ Gx,        // [1024][1200]
                    const float* __restrict__ dbih, const float* __restrict__ dbhh,
                    const float* __restrict__ fcb,
                    float* __restrict__ out) {           // [256][300]
  __shared__ __attribute__((aligned(32))) float    gates[G4];
  __shared__ __attribute__((aligned(32))) _Float16 z[640];  // [h|pad|x|pad] f16
  __shared__ __attribute__((aligned(32))) float    yv[304];
  __shared__ int flags[2];                                  // [0]=any(y>0.1) [1]=done

  const int tid  = threadIdx.x;
  const int wave = tid >> 5;
  const int lane = tid & 31;
  const int boff = (lane & 16) ? 16 : 0;   // B-fragment K sub-offset per lane half

  for (int i = tid; i < 640; i += 1024) z[i] = (_Float16)0.f;
  if (tid == 0) { flags[0] = 0; flags[1] = 0; }
  float c = 0.f;
  fill_cache(WhhA, CACHE_TILES);                 // encoder Whh slab -> LDS
  __syncthreads();

  // ---------------- encoder: 1024 steps ----------------
  for (int t = 0; t < TSEQ; ++t) {
    if (tid == 0 && t + 1 < TSEQ)
      __builtin_prefetch(Gx + (size_t)(t + 1) * G4, 0, 1);
    for (int mt = wave; mt < MT_G; mt += 32) {
      v8f acc = {};
#pragma unroll
      for (int kt = 0; kt < KT_ENC; ++kt) {
        v16h a = load_a(WhhA, mt * KT_ENC + kt, lane);
        v16h b = *(const v16h*)(z + kt * 32 + boff);
        acc = __builtin_amdgcn_wmma_f32_16x16x32_f16(false, a, false, b,
                                                     (short)0, acc, false, false);
      }
      if ((lane & 15) == 0) {
        int m0 = mt * 16 + ((lane >> 4) << 3);
        const float* gxr = Gx + (size_t)t * G4 + m0;
#pragma unroll
        for (int i = 0; i < 8; ++i) gates[m0 + i] = acc[i] + gxr[i];
      }
    }
    __syncthreads();
    if (tid < H) {
      float ig = sig(gates[tid]);
      float fg = sig(gates[H + tid]);
      float gg = tanhf(gates[2 * H + tid]);
      float og = sig(gates[3 * H + tid]);
      c = fg * c + ig * gg;
      z[tid] = (_Float16)(og * tanhf(c));
    }
    __syncthreads();
  }

  // swap the LDS slab to decoder weights
  fill_cache(decA, CACHE_TILES);
  __syncthreads();

  // ---------------- decoder: 256 autoregressive steps ----------------
  // z[320..639] is still zero => x0 = 0, matching the reference.
  for (int t = 0; t < TOUT; ++t) {
    for (int mt = wave; mt < MT_G; mt += 32) {
      v8f acc = {};
#pragma unroll
      for (int kt = 0; kt < KT_DEC; ++kt) {
        v16h a = load_a(decA, mt * KT_DEC + kt, lane);
        v16h b = *(const v16h*)(z + kt * 32 + boff);
        acc = __builtin_amdgcn_wmma_f32_16x16x32_f16(false, a, false, b,
                                                     (short)0, acc, false, false);
      }
      if ((lane & 15) == 0) {
        int m0 = mt * 16 + ((lane >> 4) << 3);
#pragma unroll
        for (int i = 0; i < 8; ++i) {
          int m = m0 + i;
          gates[m] = acc[i] + dbih[m] + dbhh[m];
        }
      }
    }
    __syncthreads();
    if (tid < H) {
      float ig = sig(gates[tid]);
      float fg = sig(gates[H + tid]);
      float gg = tanhf(gates[2 * H + tid]);
      float og = sig(gates[3 * H + tid]);
      c = fg * c + ig * gg;
      z[tid] = (_Float16)(og * tanhf(c));
    }
    if (tid == 512) flags[0] = 0;
    __syncthreads();
    // fc:  y = fc_W @ h2 + fc_b   (only h-part K tiles 0..9 of z)
    for (int mt = wave; mt < MT_FC; mt += 32) {
      v8f acc = {};
#pragma unroll
      for (int kt = 0; kt < KT_ENC; ++kt) {
        v16h a = *(const v16h*)(fcA + (((size_t)(mt * KT_ENC + kt)) << 9) + (lane << 4));
        v16h b = *(const v16h*)(z + kt * 32 + boff);
        acc = __builtin_amdgcn_wmma_f32_16x16x32_f16(false, a, false, b,
                                                     (short)0, acc, false, false);
      }
      if ((lane & 15) == 0) {
        int m0 = mt * 16 + ((lane >> 4) << 3);
#pragma unroll
        for (int i = 0; i < 8; ++i) {
          int m = m0 + i;
          yv[m] = acc[i] + (m < H ? fcb[m] : 0.f);
        }
      }
    }
    __syncthreads();
    if (tid < H) {
      float y = yv[tid];
      int done0 = flags[1];                       // done BEFORE this step's update
      out[(size_t)t * DDIM + tid] = done0 ? 0.f : y;
      if (y > 0.1f) atomicOr(&flags[0], 1);
      z[320 + tid] = (_Float16)y;                 // x_{t+1} = y_t (raw)
    }
    __syncthreads();
    if (tid == 0 && flags[0] == 0) flags[1] = 1;  // done |= all(y <= 0.1)
    __syncthreads();
  }
}

// ---------------------------------------------------------------------------
extern "C" void kernel_launch(void* const* d_in, const int* in_sizes, int n_in,
                              void* d_out, int out_size, void* d_ws, size_t ws_size,
                              hipStream_t stream) {
  (void)in_sizes; (void)n_in; (void)out_size; (void)ws_size;
  const float* X    = (const float*)d_in[0];
  const float* eWih = (const float*)d_in[1];
  const float* eWhh = (const float*)d_in[2];
  const float* ebih = (const float*)d_in[3];
  const float* ebhh = (const float*)d_in[4];
  const float* dWih = (const float*)d_in[5];
  const float* dWhh = (const float*)d_in[6];
  const float* dbih = (const float*)d_in[7];
  const float* dbhh = (const float*)d_in[8];
  const float* fcW  = (const float*)d_in[9];
  const float* fcb  = (const float*)d_in[10];

  char* ws = (char*)d_ws;
  size_t o = 0;
  _Float16* Xa   = (_Float16*)(ws + o); o += (size_t)MT_X  * KT_ENC * 1024;
  _Float16* WihB = (_Float16*)(ws + o); o += (size_t)KT_ENC * NT_W  * 1024;
  _Float16* WhhA = (_Float16*)(ws + o); o += (size_t)MT_G  * KT_ENC * 1024;
  _Float16* dA   = (_Float16*)(ws + o); o += (size_t)MT_G  * KT_DEC * 1024;
  _Float16* fA   = (_Float16*)(ws + o); o += (size_t)MT_FC * KT_ENC * 1024;
  float*    Gx   = (float*)(ws + o);    o += (size_t)TSEQ * G4 * sizeof(float);

  pack_a <<<512, 256, 0, stream>>>(X,    Xa,   MT_X,  KT_ENC, TSEQ, DDIM);
  pack_b <<<512, 256, 0, stream>>>(eWih, WihB, KT_ENC, NT_W,  G4,   DDIM);
  pack_a <<<512, 256, 0, stream>>>(eWhh, WhhA, MT_G,  KT_ENC, G4,   H);
  pack_a2<<<1024,256, 0, stream>>>(dWhh, H, dWih, DDIM, dA, MT_G, KT_DEC, G4);
  pack_a <<<256, 256, 0, stream>>>(fcW,  fA,   MT_FC, KT_ENC, DDIM, H);

  enc_input_gemm<<<(MT_X * NT_W) / 8, 256, 0, stream>>>(Xa, WihB, ebih, ebhh, Gx);
  seq_recurrence<<<1, 1024, (size_t)CACHE_TILES * 1024, stream>>>(
      WhhA, dA, fA, Gx, dbih, dbhh, fcb, (float*)d_out);
}
